// GroupedQueryAttention_11690900979796
// MI455X (gfx1250) — matrix operand order repaired and checked by reading, not compiled, not run on hardware
//
#include <hip/hip_runtime.h>
#include <math.h>

// ---------------- types ----------------
typedef __bf16 bf16;
typedef __attribute__((ext_vector_type(16))) __bf16 v16bf;
typedef __attribute__((ext_vector_type(8)))  float  v8f;

struct __attribute__((aligned(16))) U128 { unsigned int x, y, z, w; };
union FragU { v16bf v; U128 q[2]; };
union Pack8 { bf16 h[8]; U128 u; };

// ---------------- async global->LDS copy (CDNA5) with fallback ----------------
#if defined(__has_builtin)
#if __has_builtin(__builtin_amdgcn_global_load_async_to_lds_b128) && \
    __has_builtin(__builtin_amdgcn_s_wait_asynccnt)
#define HAVE_ASYNC 1
#endif
#endif
#ifndef HAVE_ASYNC
#define HAVE_ASYNC 0
#endif

#if HAVE_ASYNC
typedef __attribute__((ext_vector_type(4))) int v4i;
typedef __attribute__((address_space(1))) v4i v4i_g;
typedef __attribute__((address_space(3))) v4i v4i_l;
#endif

// copy 16 bytes global -> LDS
__device__ __forceinline__ void cp16(const bf16* g, bf16* l) {
#if HAVE_ASYNC
  v4i* gp = (v4i*)g;   // strip const, reinterpret (generic AS)
  v4i* lp = (v4i*)l;
  __builtin_amdgcn_global_load_async_to_lds_b128((v4i_g*)gp, (v4i_l*)lp, 0, 0);
#else
  *(U128*)l = *(const U128*)g;
#endif
}
__device__ __forceinline__ void cp_wait() {
#if HAVE_ASYNC
  __builtin_amdgcn_s_wait_asynccnt(0);
#endif
}

// D = A*B + C, A/B bf16 16x32 / 32x16, C/D f32 16x16
__device__ __forceinline__ v8f wmma_bf16(v16bf a, v16bf b, v8f c) {
  return __builtin_amdgcn_wmma_f32_16x16x32_bf16(false, a, false, b, (short)0, c, false, false);
}

// A fragment (16x32 bf16, M x K) from row-major LDS tile (stride in halves).
__device__ __forceinline__ v16bf load_frag_a(const bf16* tile, int strideH, int lane) {
  int m = lane & 15;
  int base = (lane < 16) ? 0 : 8;
  const bf16* p = tile + m * strideH + base;
  FragU f;
  f.q[0] = *(const U128*)(p);
  f.q[1] = *(const U128*)(p + 16);
  return f.v;
}

// B fragment (32x16 bf16, K x N) from an N-major (transposed) LDS tile tile[n][k].
__device__ __forceinline__ v16bf load_frag_b(const bf16* bt, int strideH, int lane, int nOff, int kOff) {
  int n = nOff + (lane & 15);
  int k = kOff + ((lane < 16) ? 0 : 8);
  const bf16* p = bt + n * strideH + k;
  FragU f;
  f.q[0] = *(const U128*)(p);
  f.q[1] = *(const U128*)(p + 16);
  return f.v;
}

// ---------------- conversion kernels ----------------
struct __attribute__((aligned(16))) F4 { float x, y, z, w; };

__global__ __launch_bounds__(256) void cvt4_f32_bf16(const float* __restrict__ in,
                                                     bf16* __restrict__ out, int n4) {
  int i = blockIdx.x * 256 + threadIdx.x;
  if (i < n4) {
    F4 v = ((const F4*)in)[i];
    out[i * 4 + 0] = (bf16)v.x;
    out[i * 4 + 1] = (bf16)v.y;
    out[i * 4 + 2] = (bf16)v.z;
    out[i * 4 + 3] = (bf16)v.w;
  }
}

// W [K][N] f32 -> Wt [N][K] bf16 (32x32 LDS tiles)
__global__ __launch_bounds__(256) void transpose_to_bf16(const float* __restrict__ W,
                                                         bf16* __restrict__ Wt,
                                                         int Nn, int Kk) {
  __shared__ bf16 tile[32][33];
  int bx = blockIdx.x * 32;  // N (out) base
  int by = blockIdx.y * 32;  // K (in) base
  int tx = threadIdx.x & 31, ty = threadIdx.x >> 5;  // 32 x 8
  for (int j = 0; j < 32; j += 8)
    tile[ty + j][tx] = (bf16)W[(size_t)(by + ty + j) * Nn + bx + tx];
  __syncthreads();
  for (int j = 0; j < 32; j += 8)
    Wt[(size_t)(bx + ty + j) * Kk + by + tx] = tile[tx][ty + j];
}

// ---------------- GEMM: C[M,N] = A[M,K] @ Bt[N,K]^T + bias ----------------
// Block: 128 threads (4 waves). Tile: 128 rows x 64 cols, BK=32.
// Each wave: 32 rows x 64 cols -> 8 WMMAs per K-step. Double-buffered async staging.
// MODE 0: f32 row-major [M,N];  MODE 1: bf16 [B,H,T,hd];  MODE 2: bf16 [B,H,hd,T]
template <int MODE>
__global__ __launch_bounds__(128) void gemm_bf16(const bf16* __restrict__ A,
                                                 const bf16* __restrict__ Bt,
                                                 const float* __restrict__ bias,
                                                 void* __restrict__ Out,
                                                 int M, int N, int K) {
  __shared__ bf16 At[2][128 * 32];   // 2 x 8KB
  __shared__ bf16 Bts[2][64 * 32];   // 2 x 4KB
  const int tid = threadIdx.x, lane = tid & 31, wave = tid >> 5;
  const int rowbase = blockIdx.y * 128;
  const int nbase = blockIdx.x * 64;

  v8f acc[2][4] = {};

  auto stage = [&](int buf, int kb) {
    const bf16* ap = A + (size_t)(rowbase + tid) * K + kb;
    bf16* ad = &At[buf][tid * 32];
#pragma unroll
    for (int j = 0; j < 4; j++) cp16(ap + j * 8, ad + j * 8);
    int br = tid >> 1, bc = (tid & 1) * 16;
    const bf16* bp = Bt + (size_t)(nbase + br) * K + kb + bc;
    bf16* bd = &Bts[buf][br * 32 + bc];
    cp16(bp, bd);
    cp16(bp + 8, bd + 8);
  };

  const int nIter = K / 32;
  stage(0, 0);
  for (int it = 0; it < nIter; ++it) {
    const int cur = it & 1;
    cp_wait();
    __syncthreads();
    if (it + 1 < nIter) stage(cur ^ 1, (it + 1) * 32);

    // hoist all fragment loads into distinct registers, then burst WMMAs
    v16bf bfr[4];
#pragma unroll
    for (int nf = 0; nf < 4; nf++) bfr[nf] = load_frag_b(Bts[cur], 32, lane, nf * 16, 0);
    v16bf a0 = load_frag_a(&At[cur][(wave * 32) * 32], 32, lane);
    v16bf a1 = load_frag_a(&At[cur][(wave * 32 + 16) * 32], 32, lane);
#pragma unroll
    for (int nf = 0; nf < 4; nf++) {
      acc[0][nf] = wmma_bf16(a0, bfr[nf], acc[0][nf]);
      acc[1][nf] = wmma_bf16(a1, bfr[nf], acc[1][nf]);
    }
    __syncthreads();
  }

  const int col0 = nbase + (lane & 15);
#pragma unroll
  for (int rf = 0; rf < 2; rf++) {
    const int row0 = rowbase + wave * 32 + rf * 16 + ((lane < 16) ? 0 : 8);
#pragma unroll
    for (int nf = 0; nf < 4; nf++) {
      int c = col0 + nf * 16;
      float bv = bias[c];
#pragma unroll
      for (int i = 0; i < 8; i++) {
        int r = row0 + i;
        float val = acc[rf][nf][i] + bv;
        if (MODE == 0) {
          ((float*)Out)[(size_t)r * N + c] = val;
        } else {
          int b = r >> 11, t = r & 2047, h = c >> 6, d = c & 63;
          if (MODE == 1)
            ((bf16*)Out)[(((size_t)(b * 16 + h) * 2048 + t) << 6) + d] = (bf16)val;
          else
            ((bf16*)Out)[((size_t)(b * 16 + h) * 64 + d) * 2048 + t] = (bf16)val;
        }
      }
    }
  }
}

// ---------------- flash attention (transposed scores: S^T = K.Q^T) ----------------
// Q,K: bf16 [B,H,T,64]; Vt: bf16 [B,H,64,T]; out: bf16 [B*T, 1024] (head-concat)
// Per wave: 16 queries. Scores computed transposed so a query is a LANE:
//   softmax reduction = in-lane tree + one xor-16 shuffle; m/l are lane scalars.
// Then O^T = V^T . P^T, with P^T as a single reused B fragment.
__global__ __launch_bounds__(128) void attn_kernel(const bf16* __restrict__ Q,
                                                   const bf16* __restrict__ Kmat,
                                                   const bf16* __restrict__ Vt,
                                                   bf16* __restrict__ Outb) {
  const int T = 2048;
  const int bh = blockIdx.y;   // b*16 + h
  const int qblk = blockIdx.x; // 0..31
  const int tid = threadIdx.x, lane = tid & 31, wave = tid >> 5;

  const bf16* Qh = Q + (size_t)bh * T * 64;
  const bf16* Kh = Kmat + (size_t)bh * T * 64;
  const bf16* Vh = Vt + (size_t)bh * 64 * T;

  __shared__ bf16 Ktile[2][32 * 64];   // [key][hd]  2 x 4KB
  __shared__ bf16 Vtile[2][64 * 32];   // [hd][key]  2 x 4KB
  __shared__ bf16 Ptile[4][16 * 32];   // per-wave probs [q][key]
  __shared__ bf16 Qtile[64 * 64];      // [q][hd]

  const int qbase = qblk * 64 + wave * 16;

  auto stage = [&](int buf, int kb) {
    const bf16* kp = Kh + (size_t)kb * 64 + tid * 16;
    bf16* kd = &Ktile[buf][tid * 16];
    cp16(kp, kd);
    cp16(kp + 8, kd + 8);
    int r = tid >> 1, c = (tid & 1) * 16;
    const bf16* vp = Vh + (size_t)r * T + kb + c;
    bf16* vd = &Vtile[buf][r * 32 + c];
    cp16(vp, vd);
    cp16(vp + 8, vd + 8);
  };

  stage(0, 0);

  // stage 64x64 Q block (synchronous; overlaps with async K/V fill)
  {
    int r = tid >> 1, c = (tid & 1) * 32;  // halves
    const U128* s = (const U128*)(Qh + (size_t)(qblk * 64 + r) * 64 + c);
    U128* d = (U128*)&Qtile[r * 64 + c];
    d[0] = s[0]; d[1] = s[1]; d[2] = s[2]; d[3] = s[3];
  }
  __syncthreads();
  // Q^T as B fragments: n = query (lane), k = hd
  const v16bf qb0 = load_frag_b(&Qtile[wave * 16 * 64], 64, lane, 0, 0);   // hd 0..31
  const v16bf qb1 = load_frag_b(&Qtile[wave * 16 * 64], 64, lane, 0, 32);  // hd 32..63

  float m = -INFINITY, l = 0.f;
  v8f o[4] = {};  // O^T: o[df][i] = out[d = df*16 + i + half*8][q = lane&15]

  const int qcol = qbase + (lane & 15);        // this lane's query
  const int halfOff = (lane < 16) ? 0 : 8;

  const int nIter = (qblk * 64 + 64) / 32;
  for (int it = 0; it < nIter; ++it) {
    const int kb = it * 32;
    const int cur = it & 1;
    cp_wait();
    __syncthreads();
    if (it + 1 < nIter) stage(cur ^ 1, kb + 32);

    const bool active = kb < qbase + 16;      // wave-uniform
    const bool needMask = kb + 31 > qbase;    // diagonal block only
    if (active) {
      // S^T = K . Q^T : A = K rows (key x hd), B = Q^T
      v16bf ka0 = load_frag_a(&Ktile[cur][0], 64, lane);            // keys 0-15, hd 0-31
      v16bf ka1 = load_frag_a(&Ktile[cur][32], 64, lane);           // keys 0-15, hd 32-63
      v16bf ka2 = load_frag_a(&Ktile[cur][16 * 64], 64, lane);      // keys 16-31, hd 0-31
      v16bf ka3 = load_frag_a(&Ktile[cur][16 * 64 + 32], 64, lane); // keys 16-31, hd 32-63
      v8f s0 = {}, s1 = {};
      s0 = wmma_bf16(ka0, qb0, s0);
      s1 = wmma_bf16(ka2, qb0, s1);
      s0 = wmma_bf16(ka1, qb1, s0);
      s1 = wmma_bf16(ka3, qb1, s1);

      const float scale = 0.125f;  // 1/sqrt(64)
      const int keyb = kb + halfOff;  // first key of this half's rows
      float sv[16];
      if (needMask) {
#pragma unroll
        for (int i = 0; i < 8; i++) {
          sv[i]     = (keyb + i <= qcol)      ? s0[i] * scale : -INFINITY;
          sv[8 + i] = (keyb + 16 + i <= qcol) ? s1[i] * scale : -INFINITY;
        }
      } else {
#pragma unroll
        for (int i = 0; i < 8; i++) {
          sv[i] = s0[i] * scale;
          sv[8 + i] = s1[i] * scale;
        }
      }
      // in-lane max tree over 16 values, then combine lane-halves
      float vmax = sv[0];
#pragma unroll
      for (int j = 1; j < 16; j++) vmax = fmaxf(vmax, sv[j]);
      vmax = fmaxf(vmax, __shfl_xor(vmax, 16, 32));
      float mnew = fmaxf(m, vmax);
      float corr = __expf(m - mnew);
      float rs = 0.f;
      bf16* pp = Ptile[wave] + (lane & 15) * 32 + halfOff;  // [q][key]
#pragma unroll
      for (int j = 0; j < 8; j++) {
        float p0 = __expf(sv[j] - mnew);
        float p1 = __expf(sv[8 + j] - mnew);
        rs += p0 + p1;
        pp[j] = (bf16)p0;
        pp[16 + j] = (bf16)p1;
      }
      rs += __shfl_xor(rs, 16, 32);
      l = l * corr + rs;
      m = mnew;
#pragma unroll
      for (int df = 0; df < 4; df++) o[df] = o[df] * corr;
    }
    __syncthreads();
    if (active) {
      // O^T += V^T . P^T : A = Vt rows (d x key), B = P^T (one frag, reused x4)
      v16bf pb = load_frag_b(Ptile[wave], 32, lane, 0, 0);
      v16bf va0 = load_frag_a(&Vtile[cur][0 * 16 * 32], 32, lane);
      v16bf va1 = load_frag_a(&Vtile[cur][1 * 16 * 32], 32, lane);
      v16bf va2 = load_frag_a(&Vtile[cur][2 * 16 * 32], 32, lane);
      v16bf va3 = load_frag_a(&Vtile[cur][3 * 16 * 32], 32, lane);
      o[0] = wmma_bf16(va0, pb, o[0]);
      o[1] = wmma_bf16(va1, pb, o[1]);
      o[2] = wmma_bf16(va2, pb, o[2]);
      o[3] = wmma_bf16(va3, pb, o[3]);
    }
  }

  // epilogue: out[b*2048 + q][h*64 + d]; 8 contiguous bf16 per fragment -> b128 store
  const int b = bh >> 4, h = bh & 15;
  const float rl = 1.0f / l;
  bf16* orow = Outb + ((size_t)b * 2048 + qcol) * 1024 + h * 64 + halfOff;
#pragma unroll
  for (int df = 0; df < 4; df++) {
    Pack8 pk;
#pragma unroll
    for (int i = 0; i < 8; i++) pk.h[i] = (bf16)(o[df][i] * rl);
    *(U128*)(orow + df * 16) = pk.u;
  }
}

// ---------------- launch ----------------
extern "C" void kernel_launch(void* const* d_in, const int* in_sizes, int n_in,
                              void* d_out, int out_size, void* d_ws, size_t ws_size,
                              hipStream_t stream) {
  (void)in_sizes; (void)n_in; (void)out_size; (void)ws_size;
  const int D = 1024, M = 8192;  // B*T = 4*2048
  const float* x  = (const float*)d_in[0];
  const float* Wq = (const float*)d_in[1];
  const float* bq = (const float*)d_in[2];
  const float* Wk = (const float*)d_in[3];
  const float* bk = (const float*)d_in[4];
  const float* Wv = (const float*)d_in[5];
  const float* bv = (const float*)d_in[6];
  const float* Wo = (const float*)d_in[7];
  const float* bo = (const float*)d_in[8];
  float* out = (float*)d_out;

  char* ws = (char*)d_ws;
  size_t off = 0;
  bf16* xb   = (bf16*)(ws + off); off += (size_t)M * D * 2;
  bf16* Wqt  = (bf16*)(ws + off); off += (size_t)D * D * 2;
  bf16* Wkt  = (bf16*)(ws + off); off += (size_t)D * D * 2;
  bf16* Wvt  = (bf16*)(ws + off); off += (size_t)D * D * 2;
  bf16* Wot  = (bf16*)(ws + off); off += (size_t)D * D * 2;
  bf16* Qb   = (bf16*)(ws + off); off += (size_t)M * D * 2;
  bf16* Kb   = (bf16*)(ws + off); off += (size_t)M * D * 2;
  bf16* Vtb  = (bf16*)(ws + off); off += (size_t)M * D * 2;
  bf16* attn = (bf16*)(ws + off); off += (size_t)M * D * 2;

  {
    int n4 = M * D / 4;
    cvt4_f32_bf16<<<(n4 + 255) / 256, 256, 0, stream>>>(x, xb, n4);
  }
  {
    dim3 g(D / 32, D / 32), blk(256);
    transpose_to_bf16<<<g, blk, 0, stream>>>(Wq, Wqt, D, D);
    transpose_to_bf16<<<g, blk, 0, stream>>>(Wk, Wkt, D, D);
    transpose_to_bf16<<<g, blk, 0, stream>>>(Wv, Wvt, D, D);
    transpose_to_bf16<<<g, blk, 0, stream>>>(Wo, Wot, D, D);
  }
  {
    dim3 g(D / 64, M / 128), blk(128);
    gemm_bf16<1><<<g, blk, 0, stream>>>(xb, Wqt, bq, Qb, M, D, D);
    gemm_bf16<1><<<g, blk, 0, stream>>>(xb, Wkt, bk, Kb, M, D, D);
    gemm_bf16<2><<<g, blk, 0, stream>>>(xb, Wvt, bv, Vtb, M, D, D);
  }
  attn_kernel<<<dim3(32, 64), 128, 0, stream>>>(Qb, Kb, Vtb, attn);
  gemm_bf16<0><<<dim3(D / 64, M / 128), 128, 0, stream>>>(attn, Wot, bo, out, M, D, D);
}
